// STM_5944234738226
// MI455X (gfx1250) — matrix operand order repaired and checked by reading, not compiled, not run on hardware
//
#include <hip/hip_runtime.h>

// ---------------------------------------------------------------------------
// Dims (fixed by the reference)
// B=64, IN=512, OUT=256, N=16, D=256, REL=256, OATT=128, MLPD=512, STEP=1
// ---------------------------------------------------------------------------
#define BB    64
#define INF   512
#define DD    256
#define NN    16
#define RELQ  256
#define OATT  128
#define MLPD  512
#define KTOT  65536       // D*D, K of the big GEMM
#define KSPLIT 32         // k-split groups over d (8 d-slices per WG)
#define BPB   2           // batches per workgroup in the big GEMM (B reuse)
#define EPS   1e-5f

typedef __bf16 bf16_t;
typedef bf16_t v16bf __attribute__((ext_vector_type(16)));
typedef bf16_t v8bf  __attribute__((ext_vector_type(8)));
typedef float  v8f   __attribute__((ext_vector_type(8)));

// ---------------------------------------------------------------------------
// WMMA helpers (CDNA5 v_wmma_f32_16x16x32_bf16, fp32 accumulate)
// ---------------------------------------------------------------------------
__device__ __forceinline__ v8f wmma_bf16(v16bf a, v16bf b, v8f c) {
  return __builtin_amdgcn_wmma_f32_16x16x32_bf16(false, a, false, b,
                                                 (short)0, c, false, false);
}

// A-matrix 16x32 bf16 fragment from an LDS row (row = matrix row m).
// Lane half h: elements 0..15 -> K=kb+16h+e
__device__ __forceinline__ v16bf load_afrag(const bf16_t* row, int kb, int h) {
  v8bf lo = *(const v8bf*)(row + kb + 16 * h);
  v8bf hi = *(const v8bf*)(row + kb + 16 * h + 8);
  v16bf a;
#pragma unroll
  for (int i = 0; i < 8; ++i) { a[i] = lo[i]; a[i + 8] = hi[i]; }
  return a;
}

// B-matrix 32x16 bf16 fragment: pre-swizzled in memory so each lane's 16
// elements (col r = 16*tile + lane%16, K = 32*chunk + 16*(lane/16) + e) are
// contiguous (32B -> two b128 loads).
__device__ __forceinline__ v16bf load_bfrag(const bf16_t* g) {
  const v8bf* p = (const v8bf*)g;
  v8bf lo = p[0], hi = p[1];
  v16bf b;
#pragma unroll
  for (int i = 0; i < 8; ++i) { b[i] = lo[i]; b[i + 8] = hi[i]; }
  return b;
}

// ---------------------------------------------------------------------------
// PW: swizzle an fp32 row-major weight [rows=16*ncolT, Ktot] into the
// WMMA-B fragment-ready bf16 layout: [colTile][chunk][lane][16 elems].
// ---------------------------------------------------------------------------
__global__ __launch_bounds__(256) void k_swz(const float* __restrict__ W,
                                             bf16_t* __restrict__ out,
                                             int nchunk, int ktot, long total) {
  long gid = (long)blockIdx.x * 256 + threadIdx.x;
  if (gid >= total) return;
  int  e    = (int)(gid & 15);
  int  lane = (int)((gid >> 4) & 31);
  long idx2 = gid >> 9;
  int  c    = (int)(idx2 % nchunk);
  int  w    = (int)(idx2 / nchunk);
  int  r    = w * 16 + (lane & 15);
  int  K    = c * 32 + ((lane >> 4) << 4) + e;
  out[gid] = (bf16_t)W[(size_t)r * ktot + K];
}

// ---------------------------------------------------------------------------
// P0: per-batch small projections: c, c2, softmax c3, gate_in
// ---------------------------------------------------------------------------
__global__ __launch_bounds__(256) void k_proj(
    const float* __restrict__ x,
    const float* __restrict__ W1, const float* __restrict__ b1,
    const float* __restrict__ W2, const float* __restrict__ b2,
    const float* __restrict__ W3, const float* __restrict__ b3,
    const float* __restrict__ Wig, const float* __restrict__ big,
    float* __restrict__ c, float* __restrict__ c2, float* __restrict__ c3,
    float* __restrict__ gate_in) {
  int b = blockIdx.x, t = threadIdx.x;
  __shared__ float lx[INF];
  __shared__ float lc[DD];
  __shared__ float lraw[NN];
  lx[t]       = x[b * INF + t];
  lx[t + 256] = x[b * INF + 256 + t];
  __syncthreads();
  {
    float s1 = b1[t], s2 = b2[t];
    const float* w1 = W1 + (size_t)t * INF;
    const float* w2 = W2 + (size_t)t * INF;
    for (int i = 0; i < INF; ++i) { s1 += lx[i] * w1[i]; s2 += lx[i] * w2[i]; }
    c[b * DD + t] = s1; lc[t] = s1; c2[b * DD + t] = s2;
  }
  if (t < NN) {
    float s = b3[t];
    const float* w = W3 + (size_t)t * INF;
    for (int i = 0; i < INF; ++i) s += lx[i] * w[i];
    lraw[t] = s;
  }
  __syncthreads();
  if (t == 0) {
    float mx = lraw[0];
    for (int i = 1; i < NN; ++i) mx = fmaxf(mx, lraw[i]);
    float sum = 0.f;
    for (int i = 0; i < NN; ++i) { lraw[i] = __expf(lraw[i] - mx); sum += lraw[i]; }
    float inv = 1.f / sum;
    for (int i = 0; i < NN; ++i) c3[b * NN + i] = lraw[i] * inv;
  }
  // gate_in = c @ W_ig.T + b_ig   (512 outputs)
#pragma unroll
  for (int p = 0; p < 2; ++p) {
    int o = t + p * 256;
    float s = big[o];
    const float* w = Wig + (size_t)o * DD;
    for (int i = 0; i < DD; ++i) s += lc[i] * w[i];
    gate_in[b * 512 + o] = s;
  }
}

// ---------------------------------------------------------------------------
// P1: c4 partials over n:  part[b,n,f] = sum_d c2[b,d]*rel_mem[b,n,d,f]
// (one streaming pass over rel_mem, fully coalesced)
// ---------------------------------------------------------------------------
__global__ __launch_bounds__(256) void k_c4part(const float* __restrict__ rel_mem,
                                                const float* __restrict__ c2,
                                                float* __restrict__ part) {
  int b = blockIdx.x >> 4, n = blockIdx.x & 15, t = threadIdx.x;
  __shared__ float lc2[DD];
  lc2[t] = c2[b * DD + t];
  __syncthreads();
  const float* rm = rel_mem + (size_t)(b * NN + n) * KTOT;
  float acc = 0.f;
  for (int d = 0; d < DD; ++d) acc += lc2[d] * rm[(size_t)d * DD + t];
  part[(size_t)(b * NN + n) * DD + t] = acc;
}

// P1b: c4[b,f] = sum_n c3[b,n]*part[b,n,f]   (deterministic reduction)
__global__ __launch_bounds__(256) void k_c4(const float* __restrict__ part,
                                            const float* __restrict__ c3,
                                            float* __restrict__ c4) {
  int b = blockIdx.x, t = threadIdx.x;
  float acc = 0.f;
#pragma unroll
  for (int n = 0; n < NN; ++n)
    acc += c3[b * NN + n] * part[(size_t)(b * NN + n) * DD + t];
  c4[b * DD + t] = acc;
}

// ---------------------------------------------------------------------------
// P2: per (b, 16-row d-tile): WMMA gate_mem = tanh(item)@W_mg.T, build
// Aq = i*tanh(X) + f*item + alpha2*X2, fused qkv projection + LN,
// scatter into q,k,v [B,16,256].
// ---------------------------------------------------------------------------
__global__ __launch_bounds__(512) void k_gate_qkv(
    const float* __restrict__ item_mem, const bf16_t* __restrict__ Wmg_swz,
    const float* __restrict__ b_mg, const float* __restrict__ gate_in,
    const float* __restrict__ c, const float* __restrict__ c2,
    const float* __restrict__ c4, const float* __restrict__ in_bias,
    const float* __restrict__ fg_bias, const float* __restrict__ alpha2p,
    const float* __restrict__ W_qkv, const float* __restrict__ b_qkv,
    const float* __restrict__ g_qkv, const float* __restrict__ be_qkv,
    float* __restrict__ qo, float* __restrict__ ko, float* __restrict__ vo) {
  int b = blockIdx.x >> 4, dd = blockIdx.x & 15, d0 = dd * 16;
  int t = threadIdx.x, lane = t & 31, wave = t >> 5;

  __shared__ __align__(16) float  ld_item[16][DD];     // raw item tile, reused as Aq
  __shared__ __align__(16) bf16_t ld_a[16][272];       // tanh(item) bf16 (padded)
  __shared__ float ld_gate[16][512];                   // gate GEMM result
  __shared__ float ld_y[16][48];

  // stage item tile + tanh->bf16
  {
    const float* it = item_mem + ((size_t)b * DD + d0) * DD;
    for (int i = t; i < 1024; i += 512) {
      float4 xv = ((const float4*)it)[i];
      ((float4*)&ld_item[0][0])[i] = xv;
      int m = i >> 6, f = (i & 63) * 4;
      ld_a[m][f + 0] = (bf16_t)tanhf(xv.x);
      ld_a[m][f + 1] = (bf16_t)tanhf(xv.y);
      ld_a[m][f + 2] = (bf16_t)tanhf(xv.z);
      ld_a[m][f + 3] = (bf16_t)tanhf(xv.w);
    }
  }
  __syncthreads();

  // gate GEMM: rows = d-local (16), cols = 512 gate outputs, K=256
  v8f acc0 = {}, acc1 = {};
  const int h = lane >> 4, colr = lane & 15;
#pragma unroll
  for (int cc = 0; cc < 8; ++cc) {
    v16bf a  = load_afrag(&ld_a[colr][0], cc * 32, h);
    v16bf b0 = load_bfrag(Wmg_swz + ((size_t)((wave * 2 + 0) * 8 + cc) * 32 + lane) * 16);
    v16bf b1 = load_bfrag(Wmg_swz + ((size_t)((wave * 2 + 1) * 8 + cc) * 32 + lane) * 16);
    acc0 = wmma_bf16(a, b0, acc0);
    acc1 = wmma_bf16(a, b1, acc1);
  }
#pragma unroll
  for (int v = 0; v < 8; ++v) {
    int m = v + 8 * h;
    ld_gate[m][wave * 32 + colr]      = acc0[v];
    ld_gate[m][wave * 32 + 16 + colr] = acc1[v];
  }
  __syncthreads();

  // Aq = sigmoid(i)*tanh(X) + sigmoid(f)*item + alpha2*X2  (in place of ld_item)
  {
    float ib = in_bias[0], fb = fg_bias[0], a2 = alpha2p[0];
    int m = t >> 5, f0 = (t & 31) * 8;
    float cd  = c[b * DD + d0 + m];
    float c4d = c4[b * DD + d0 + m];
#pragma unroll
    for (int e = 0; e < 8; ++e) {
      int f = f0 + e;
      float gi = ld_gate[m][f]       + b_mg[f]       + gate_in[b * 512 + f]       + ib;
      float gf = ld_gate[m][f + 256] + b_mg[f + 256] + gate_in[b * 512 + f + 256] + fb;
      float sg_i = 1.f / (1.f + __expf(-gi));
      float sg_f = 1.f / (1.f + __expf(-gf));
      float X  = cd  * c[b * DD + f];
      float X2 = c4d * c2[b * DD + f];
      float itv = ld_item[m][f];
      ld_item[m][f] = sg_i * tanhf(X) + sg_f * itv + a2 * X2;  // now Aq
    }
  }
  __syncthreads();

  // qkv = Aq @ W_qkv.T + b_qkv   (48 outputs per row)
  for (int idx = t; idx < 768; idx += 512) {
    int m = idx & 15, o = idx >> 4;
    const float* wrow = W_qkv + (size_t)o * DD;
    float s = b_qkv[o];
    for (int f = 0; f < DD; ++f) s += ld_item[m][f] * wrow[f];
    ld_y[m][o] = s;
  }
  __syncthreads();

  // LayerNorm over 48 + scatter to q,k,v
  if (t < 16) {
    int m = t, dg = d0 + m;
    float mean = 0.f;
    for (int o = 0; o < 48; ++o) mean += ld_y[m][o];
    mean *= (1.f / 48.f);
    float var = 0.f;
    for (int o = 0; o < 48; ++o) { float dv = ld_y[m][o] - mean; var += dv * dv; }
    var *= (1.f / 48.f);
    float inv = rsqrtf(var + EPS);
    for (int o = 0; o < 48; ++o) {
      float yv = (ld_y[m][o] - mean) * inv * g_qkv[o] + be_qkv[o];
      if (o < 16)      qo[((size_t)b * NN + o)        * DD + dg] = yv;
      else if (o < 32) ko[((size_t)b * NN + (o - 16)) * DD + dg] = yv;
      else             vo[((size_t)b * NN + (o - 32)) * DD + dg] = yv;
    }
  }
}

// ---------------------------------------------------------------------------
// P3: the big fused GEMM.  Per WG = (batch-pair, k-split of 8 d-slices):
// for each d, build A_bb[n][f] = alpha1*rel_mem[b,n,d,f] + sum_j T*v in LDS
// (bf16) for BOTH batches, then each of the 16 waves loads one B fragment of
// the L2-resident pre-swizzled W_rel and feeds TWO v_wmma_f32_16x16x32_bf16
// (one per batch) -> 2x B-fragment reuse, halving L2 traffic (~2GB -> ~1GB).
// Full-K partials are written deterministically to rpart[ks][b][n*256+r].
// ---------------------------------------------------------------------------
__global__ __launch_bounds__(512) void k_relgemm(
    const float* __restrict__ rel_mem, const float* __restrict__ q,
    const float* __restrict__ k, const float* __restrict__ v,
    const bf16_t* __restrict__ Wswz, const float* __restrict__ alpha1p,
    float* __restrict__ rpart) {
  const int bp = blockIdx.x >> 5;      // 0..31 batch pair
  const int ks = blockIdx.x & 31;      // 0..31 k-split
  const int d0 = ks * 8;
  const int t = threadIdx.x, lane = t & 31, wave = t >> 5;
  const float alpha1 = alpha1p[0];

  __shared__ __align__(16) float  ld_v[BPB][NN][DD];   // 32KB
  __shared__ float ld_q[BPB][NN][8];
  __shared__ float ld_k[BPB][NN][8];
  __shared__ float ld_T[BPB][NN][NN];
  __shared__ __align__(16) bf16_t ld_a[BPB][NN][272];  // padded rows

  // stage v for both batches (b-consecutive rows are contiguous in global)
  {
    const float4* src = (const float4*)(v + (size_t)bp * BPB * NN * DD);
    float4* dst = (float4*)&ld_v[0][0][0];
    for (int i = t; i < (BPB * NN * DD) / 4; i += 512) dst[i] = src[i];
  }
  // stage q,k restricted to this WG's 8 d-slices
  if (t < 256) {
    int bb = t >> 7, n = (t >> 3) & 15, dl = t & 7;
    size_t base = ((size_t)(bp * BPB + bb) * NN + n) * DD + d0 + dl;
    ld_q[bb][n][dl] = q[base];
    ld_k[bb][n][dl] = k[base];
  }
  __syncthreads();

  v8f acc[BPB] = {};
  const int h = lane >> 4;

  for (int dl = 0; dl < 8; ++dl) {
    const int d = d0 + dl;
    // T_bb[n][j] = tanh(q_nd * k_jd)   (512 threads cover 2*256)
    {
      int bb = t >> 8, n = (t >> 4) & 15, j = t & 15;
      ld_T[bb][n][j] = tanhf(ld_q[bb][n][dl] * ld_k[bb][j][dl]);
    }
    __syncthreads();

    // build both A tiles (bf16): thread -> (n = t/32, f = (t%32)*8 .. +8)
#pragma unroll
    for (int bb = 0; bb < BPB; ++bb) {
      int n = t >> 5, f0 = (t & 31) * 8;
      const float* src = rel_mem +
          ((size_t)(bp * BPB + bb) * NN + n) * KTOT + (size_t)d * DD + f0;
      __builtin_prefetch(src + DD, 0, 1);   // next d slice -> global_prefetch_b8
      float4 r0 = ((const float4*)src)[0];
      float4 r1 = ((const float4*)src)[1];
      float rm8[8] = {r0.x, r0.y, r0.z, r0.w, r1.x, r1.y, r1.z, r1.w};
      float tn[NN];
#pragma unroll
      for (int j = 0; j < NN; ++j) tn[j] = ld_T[bb][n][j];
#pragma unroll
      for (int e = 0; e < 8; ++e) {
        int f = f0 + e;
        float s = alpha1 * rm8[e];
#pragma unroll
        for (int j = 0; j < NN; ++j) s += tn[j] * ld_v[bb][j][f];
        ld_a[bb][n][f] = (bf16_t)s;
      }
    }
    __syncthreads();

    // 8 WMMA K-chunks over this d's 256 columns; one B frag -> two WMMAs
#pragma unroll
    for (int cc = 0; cc < 8; ++cc) {
      const bf16_t* bg =
          Wswz + ((size_t)((size_t)wave * 2048 + (size_t)d * 8 + cc) * 32 + lane) * 16;
      v16bf bb0 = load_bfrag(bg);
      v16bf a0 = load_afrag(&ld_a[0][lane & 15][0], cc * 32, h);
      acc[0] = wmma_bf16(a0, bb0, acc[0]);
      v16bf a1 = load_afrag(&ld_a[1][lane & 15][0], cc * 32, h);
      acc[1] = wmma_bf16(a1, bb0, acc[1]);
    }
    __syncthreads();
  }

  // store deterministic partial tiles: rows n, cols r = 16*wave + lane%16
#pragma unroll
  for (int bb = 0; bb < BPB; ++bb) {
    float* out = rpart + (size_t)(ks * BB + bp * BPB + bb) * (NN * RELQ);
#pragma unroll
    for (int vg = 0; vg < 8; ++vg) {
      int m = vg + 8 * h;
      out[m * RELQ + wave * 16 + (lane & 15)] = acc[bb][vg];
    }
  }
}

// ---------------------------------------------------------------------------
// P4: reduce k-split partials (+b_rel), logits + LN, 3-layer MLP head.
// ---------------------------------------------------------------------------
__global__ __launch_bounds__(256) void k_head(
    const float* __restrict__ rpart, const float* __restrict__ b_rel,
    const float* __restrict__ W_rel3, const float* __restrict__ b_rel3,
    const float* __restrict__ g_ln2, const float* __restrict__ be_ln2,
    const float* __restrict__ W_m1, const float* __restrict__ b_m1,
    const float* __restrict__ W_m2, const float* __restrict__ b_m2,
    const float* __restrict__ W_o, const float* __restrict__ b_o,
    float* __restrict__ outp) {
  int b = blockIdx.x, t = threadIdx.x;
  __shared__ float lr[NN * RELQ];     // 4096
  __shared__ float llog[OATT];
  __shared__ float lh1[MLPD];
  __shared__ float lh2[MLPD];
  __shared__ float lstat[2];

  for (int i = t; i < NN * RELQ; i += 256) {
    float s = b_rel[i & (RELQ - 1)];
#pragma unroll
    for (int ksp = 0; ksp < KSPLIT; ++ksp)
      s += rpart[(size_t)(ksp * BB + b) * (NN * RELQ) + i];
    lr[i] = s;
  }
  __syncthreads();
  if (t < OATT) {
    const float* w = W_rel3 + (size_t)t * (NN * RELQ);
    float s = b_rel3[t];
    for (int i = 0; i < NN * RELQ; ++i) s += lr[i] * w[i];
    llog[t] = s;
  }
  __syncthreads();
  if (t == 0) {
    float mean = 0.f;
    for (int i = 0; i < OATT; ++i) mean += llog[i];
    mean *= (1.f / OATT);
    float var = 0.f;
    for (int i = 0; i < OATT; ++i) { float dv = llog[i] - mean; var += dv * dv; }
    var *= (1.f / OATT);
    lstat[0] = mean; lstat[1] = rsqrtf(var + EPS);
  }
  __syncthreads();
  if (t < OATT) llog[t] = (llog[t] - lstat[0]) * lstat[1] * g_ln2[t] + be_ln2[t];
  __syncthreads();
#pragma unroll
  for (int p = 0; p < 2; ++p) {
    int o = t + p * 256;
    float s = b_m1[o];
    const float* w = W_m1 + (size_t)o * OATT;
    for (int i = 0; i < OATT; ++i) s += llog[i] * w[i];
    lh1[o] = fmaxf(s, 0.f);
  }
  __syncthreads();
#pragma unroll
  for (int p = 0; p < 2; ++p) {
    int o = t + p * 256;
    float s = b_m2[o];
    const float* w = W_m2 + (size_t)o * MLPD;
    for (int i = 0; i < MLPD; ++i) s += lh1[i] * w[i];
    lh2[o] = fmaxf(s, 0.f);
  }
  __syncthreads();
  {
    float s = b_o[t];
    const float* w = W_o + (size_t)t * MLPD;
    for (int i = 0; i < MLPD; ++i) s += lh2[i] * w[i];
    outp[b * 256 + t] = s;
  }
}

// ---------------------------------------------------------------------------
// Launch
// ---------------------------------------------------------------------------
extern "C" void kernel_launch(void* const* d_in, const int* in_sizes, int n_in,
                              void* d_out, int out_size, void* d_ws, size_t ws_size,
                              hipStream_t stream) {
  (void)in_sizes; (void)n_in; (void)out_size; (void)ws_size;
  const float* x        = (const float*)d_in[0];
  const float* item_mem = (const float*)d_in[1];
  const float* rel_mem  = (const float*)d_in[2];
  const float* W_in1 = (const float*)d_in[3];  const float* b_in1 = (const float*)d_in[4];
  const float* W_in2 = (const float*)d_in[5];  const float* b_in2 = (const float*)d_in[6];
  const float* W_in3 = (const float*)d_in[7];  const float* b_in3 = (const float*)d_in[8];
  const float* W_ig  = (const float*)d_in[9];  const float* b_ig  = (const float*)d_in[10];
  const float* W_mg  = (const float*)d_in[11]; const float* b_mg  = (const float*)d_in[12];
  const float* input_bias  = (const float*)d_in[13];
  const float* forget_bias = (const float*)d_in[14];
  const float* W_qkv = (const float*)d_in[15]; const float* b_qkv = (const float*)d_in[16];
  const float* g_qkv = (const float*)d_in[17]; const float* be_qkv = (const float*)d_in[18];
  const float* alpha1 = (const float*)d_in[19];
  const float* alpha2 = (const float*)d_in[20];
  // d_in[21] alpha3, d_in[22] W_rel2, d_in[23] b_rel2: dead with STEP=1
  const float* W_rel  = (const float*)d_in[24]; const float* b_rel  = (const float*)d_in[25];
  const float* W_rel3 = (const float*)d_in[26]; const float* b_rel3 = (const float*)d_in[27];
  const float* g_ln2  = (const float*)d_in[28]; const float* be_ln2 = (const float*)d_in[29];
  const float* W_m1 = (const float*)d_in[30]; const float* b_m1 = (const float*)d_in[31];
  const float* W_m2 = (const float*)d_in[32]; const float* b_m2 = (const float*)d_in[33];
  const float* W_o  = (const float*)d_in[34]; const float* b_o  = (const float*)d_in[35];

  char* ws = (char*)d_ws;
  size_t off = 0;
  auto alloc = [&](size_t bytes) {
    size_t o = off;
    off += (bytes + 255) & ~(size_t)255;
    return o;
  };
  const long WREL_TOT = (long)16 * 2048 * 32 * 16;  // 16,777,216 bf16 elems
  const long WMG_TOT  = (long)32 * 8 * 32 * 16;     // 131,072 bf16 elems
  bf16_t* wrel_bf = (bf16_t*)(ws + alloc((size_t)WREL_TOT * 2));
  bf16_t* wmg_bf  = (bf16_t*)(ws + alloc((size_t)WMG_TOT * 2));
  float* c_buf   = (float*)(ws + alloc((size_t)BB * DD * 4));
  float* c2_buf  = (float*)(ws + alloc((size_t)BB * DD * 4));
  float* c3_buf  = (float*)(ws + alloc((size_t)BB * NN * 4));
  float* c4_buf  = (float*)(ws + alloc((size_t)BB * DD * 4));
  float* gi_buf  = (float*)(ws + alloc((size_t)BB * 512 * 4));
  float* c4part  = (float*)(ws + alloc((size_t)BB * NN * DD * 4));
  float* q_buf   = (float*)(ws + alloc((size_t)BB * NN * DD * 4));
  float* k_buf   = (float*)(ws + alloc((size_t)BB * NN * DD * 4));
  float* v_buf   = (float*)(ws + alloc((size_t)BB * NN * DD * 4));
  float* rpart   = (float*)(ws + alloc((size_t)KSPLIT * BB * NN * RELQ * 4));

  k_swz<<<(int)((WREL_TOT + 255) / 256), 256, 0, stream>>>(W_rel, wrel_bf, 2048, KTOT, WREL_TOT);
  k_swz<<<(int)((WMG_TOT + 255) / 256), 256, 0, stream>>>(W_mg, wmg_bf, 8, DD, WMG_TOT);
  k_proj<<<BB, 256, 0, stream>>>(x, W_in1, b_in1, W_in2, b_in2, W_in3, b_in3,
                                 W_ig, b_ig, c_buf, c2_buf, c3_buf, gi_buf);
  k_c4part<<<BB * NN, 256, 0, stream>>>(rel_mem, c2_buf, c4part);
  k_c4<<<BB, 256, 0, stream>>>(c4part, c3_buf, c4_buf);
  k_gate_qkv<<<BB * 16, 512, 0, stream>>>(item_mem, wmg_bf, b_mg, gi_buf, c_buf,
                                          c2_buf, c4_buf, input_bias, forget_bias,
                                          alpha2, W_qkv, b_qkv, g_qkv, be_qkv,
                                          q_buf, k_buf, v_buf);
  k_relgemm<<<(BB / BPB) * KSPLIT, 512, 0, stream>>>(rel_mem, q_buf, k_buf, v_buf,
                                                     wrel_bf, alpha1, rpart);
  k_head<<<BB, 256, 0, stream>>>(rpart, b_rel, W_rel3, b_rel3, g_ln2, be_ln2,
                                 W_m1, b_m1, W_m2, b_m2, W_o, b_o, (float*)d_out);
}
